// FewShotSeg_64201171140876
// MI455X (gfx1250) — matrix operand order repaired and checked by reading, not compiled, not run on hardware
//
#include <hip/hip_runtime.h>

// ---------------------------------------------------------------------------
// Types / helpers
// ---------------------------------------------------------------------------
typedef __attribute__((ext_vector_type(16))) __bf16 bf16x16;
typedef __attribute__((ext_vector_type(8)))  __bf16 bf16x8;
typedef __attribute__((ext_vector_type(8)))  float  f32x8;
typedef __attribute__((ext_vector_type(4)))  int    v4i;
typedef unsigned short u16;

#if defined(__has_builtin)
#  if __has_builtin(__builtin_amdgcn_global_load_async_to_lds_b128)
#    define HAVE_ASYNC 1
#  endif
#endif

#ifdef HAVE_ASYNC
#  if __has_builtin(__builtin_amdgcn_s_wait_asynccnt)
#    define WAITA4() __builtin_amdgcn_s_wait_asynccnt(4)
#    define WAITA0() __builtin_amdgcn_s_wait_asynccnt(0)
#  else
#    define WAITA4() asm volatile("s_wait_asynccnt 0x4" ::: "memory")
#    define WAITA0() asm volatile("s_wait_asynccnt 0x0" ::: "memory")
#  endif
#else
#  define WAITA4()
#  define WAITA0()
#endif

__device__ __forceinline__ u16 f32_bf16(float f) {
    union { float f; unsigned u; } c; c.f = f;
    unsigned u = c.u;
    u += 0x7FFFu + ((u >> 16) & 1u);   // RNE
    return (u16)(u >> 16);
}

// ---------------------------------------------------------------------------
// WMMA GEMM:  out[b][m][n] = alpha * sum_k A[m][k]*B[n][k]  (+bias)
// A: bf16 row-major M x K (lda), B: bf16 row-major N x K (ldb) -- K contiguous.
// 256 threads = 8 waves (4 waveM x 2 waveN); WG tile 128x128; per-wave 32x64
// (2x4 of 16x16 accumulators -> 8 v_wmma per K-step of 32).
// LDS double-buffered; async global->LDS staging when available.
// M % 128 == 0 and K % 32 == 0 required; N guarded.
// biasMode: 0 none, 1 bias[n], 2 bias[m].  Writes outH (bf16) if set else outF.
// ---------------------------------------------------------------------------
#define GTM 128
#define GTN 128
#define GTK 32
#define LDSP 40
#define TILEU (128 * LDSP)     // ushorts per tile (5120 -> 10240 B)

__device__ __forceinline__ void stage16(const u16* g, u16* l)
{
#ifdef HAVE_ASYNC
    __builtin_amdgcn_global_load_async_to_lds_b128((v4i*)g, (v4i*)l, 0, 0);
#else
    *(uint4*)l = *(const uint4*)g;
#endif
}

__device__ __forceinline__ void stage_tiles(const u16* __restrict__ A,
                                            const u16* __restrict__ B,
                                            u16* Al, u16* Bl,
                                            int tid, int m0, int n0, int k0,
                                            int N, int lda, int ldb)
{
    // A tile: 128 m x 32 k  = 512 x 16B chunks, 2 per thread (always in range)
    #pragma unroll
    for (int e = tid; e < 512; e += 256) {
        int m = e >> 2, kc = (e & 3) << 3;
        stage16(A + (long)(m0 + m) * lda + (k0 + kc), Al + m * LDSP + kc);
    }
    // B tile: 128 n x 32 k; clamp OOB rows (their columns are never stored)
    #pragma unroll
    for (int e = tid; e < 512; e += 256) {
        int n = e >> 2, kc = (e & 3) << 3;
        int nn = n0 + n; if (nn > N - 1) nn = N - 1;
        stage16(B + (long)nn * ldb + (k0 + kc), Bl + n * LDSP + kc);
    }
}

__launch_bounds__(256)
__global__ void wmma_gemm(const u16* __restrict__ Ag, const u16* __restrict__ Bg,
                          const float* __restrict__ bias, int biasMode,
                          float* __restrict__ outF, u16* __restrict__ outH,
                          int M, int N, int K, int lda, int ldb,
                          long bsA, long bsB, long bsO, float alpha)
{
    __shared__ u16 lds[4 * TILEU];          // {A0,B0,A1,B1} = 40 KB

    const int tid   = threadIdx.x;
    const int lane  = tid & 31, wid = tid >> 5;
    const int waveM = wid >> 1, waveN = wid & 1;
    const int lmod  = lane & 15, lhalf = lane >> 4;
    const int m0 = blockIdx.x * GTM, n0 = blockIdx.y * GTN;
    const long bz = blockIdx.z;

    const u16* A = Ag + bz * bsA;
    const u16* B = Bg + bz * bsB;

    f32x8 acc[2][4] = {};
    const int nsteps = K / GTK;

    stage_tiles(A, B, lds, lds + TILEU, tid, m0, n0, 0, N, lda, ldb);

    for (int s = 0; s < nsteps; ++s) {
        const int cur = s & 1;
        if (s + 1 < nsteps) {
            u16* nb = lds + (1 - cur) * 2 * TILEU;
            stage_tiles(A, B, nb, nb + TILEU, tid, m0, n0, (s + 1) * GTK, N, lda, ldb);
            WAITA4();                        // current step's 4 async ops done
        } else {
            WAITA0();
        }
        __syncthreads();

        const u16* Al = lds + cur * 2 * TILEU;
        const u16* Bl = Al + TILEU;

        bf16x16 fa[2], fb[4];
        #pragma unroll
        for (int sm = 0; sm < 2; ++sm) {
            const u16* ap = &Al[(waveM * 32 + sm * 16 + lmod) * LDSP];
            bf16x8 lo = *(const bf16x8*)(ap + lhalf * 8);       // K = lhalf*8 + 0..7
            bf16x8 hi = *(const bf16x8*)(ap + 16 + lhalf * 8);  // K = 16 + lhalf*8 + 0..7
            #pragma unroll
            for (int i = 0; i < 8; ++i) { fa[sm][i] = lo[i]; fa[sm][8 + i] = hi[i]; }
        }
        #pragma unroll
        for (int sn = 0; sn < 4; ++sn) {
            const u16* bp = &Bl[(waveN * 64 + sn * 16 + lmod) * LDSP + lhalf * 8];
            bf16x8 lo = *(const bf16x8*)(bp);                   // K = lhalf*8 + 0..7
            bf16x8 hi = *(const bf16x8*)(bp + 16);              // K = 16 + lhalf*8 + 0..7
            #pragma unroll
            for (int i = 0; i < 8; ++i) { fb[sn][i] = lo[i]; fb[sn][8 + i] = hi[i]; }
        }
        #pragma unroll
        for (int sm = 0; sm < 2; ++sm)
            #pragma unroll
            for (int sn = 0; sn < 4; ++sn)
                acc[sm][sn] = __builtin_amdgcn_wmma_f32_16x16x32_bf16(
                    false, fa[sm], false, fb[sn], (short)0, acc[sm][sn], false, false);
        __syncthreads();
    }

    // epilogue (C/D layout: VGPR r -> M = r + lhalf*8, N = lmod)
    float* OF = outF + bz * bsO;
    u16*   OH = outH + bz * bsO;
    #pragma unroll
    for (int sm = 0; sm < 2; ++sm) {
        #pragma unroll
        for (int sn = 0; sn < 4; ++sn) {
            int n = n0 + waveN * 64 + sn * 16 + lmod;
            if (n >= N) continue;
            float bn = (biasMode == 1) ? bias[n] : 0.0f;
            int mb = m0 + waveM * 32 + sm * 16 + lhalf * 8;
            #pragma unroll
            for (int r = 0; r < 8; ++r) {
                float bm = (biasMode == 2) ? bias[mb + r] : 0.0f;
                float val = alpha * acc[sm][sn][r] + bn + bm;
                if (outH) OH[(long)(mb + r) * N + n] = f32_bf16(val);
                else      OF[(long)(mb + r) * N + n] = val;
            }
        }
    }
}

// ---------------------------------------------------------------------------
// Pack f32 (R x C) -> bf16 transposed (C x R), tiled.  grid(C/32, R/32, B)
// ---------------------------------------------------------------------------
__launch_bounds__(256)
__global__ void pack_transpose(const float* __restrict__ in, u16* __restrict__ out,
                               int R, int Cl)
{
    __shared__ float t[32][33];
    long b = blockIdx.z;
    in  += b * (long)R * Cl;
    out += b * (long)R * Cl;
    int c0 = blockIdx.x * 32, r0 = blockIdx.y * 32;
    int tx = threadIdx.x & 31, ty = threadIdx.x >> 5;
    for (int i = ty; i < 32; i += 8) t[i][tx] = in[(long)(r0 + i) * Cl + c0 + tx];
    __syncthreads();
    for (int i = ty; i < 32; i += 8)
        out[(long)(c0 + i) * R + r0 + tx] = f32_bf16(t[tx][i]);
}

// bf16 (R x C) -> bf16 (C x R), tiled.  grid(C/32, R/32, B)
__launch_bounds__(256)
__global__ void transpose_bf16(const u16* __restrict__ in, u16* __restrict__ out,
                               int R, int Cl)
{
    __shared__ u16 t[32][33];
    long b = blockIdx.z;
    in  += b * (long)R * Cl;
    out += b * (long)R * Cl;
    int c0 = blockIdx.x * 32, r0 = blockIdx.y * 32;
    int tx = threadIdx.x & 31, ty = threadIdx.x >> 5;
    for (int i = ty; i < 32; i += 8) t[i][tx] = in[(long)(r0 + i) * Cl + c0 + tx];
    __syncthreads();
    for (int i = ty; i < 32; i += 8) out[(long)(c0 + i) * R + r0 + tx] = t[tx][i];
}

// straight f32 -> bf16 copy (weights)
__launch_bounds__(256)
__global__ void pack_copy(const float* __restrict__ in, u16* __restrict__ out, long n)
{
    long i = (long)blockIdx.x * 256 + threadIdx.x;
    if (i < n) out[i] = f32_bf16(in[i]);
}

// ---------------------------------------------------------------------------
// Row softmax: read f32 logits, write normalized bf16.  grid(rows, B), N<=3072
// ---------------------------------------------------------------------------
__launch_bounds__(256)
__global__ void softmax_rows_bf16(const float* __restrict__ S, u16* __restrict__ P, int N)
{
    long base = ((long)blockIdx.y * gridDim.x + blockIdx.x) * (long)N;
    const float* row = S + base;
    u16* prow = P + base;
    __shared__ float red[256];
    int t = threadIdx.x;

    float v[12];
    int cnt = 0;
    float mx = -1e30f;
    for (int i = t; i < N; i += 256) { float x = row[i]; v[cnt++] = x; mx = fmaxf(mx, x); }
    red[t] = mx; __syncthreads();
    for (int s = 128; s > 0; s >>= 1) { if (t < s) red[t] = fmaxf(red[t], red[t + s]); __syncthreads(); }
    mx = red[0]; __syncthreads();

    float sum = 0.f;
    for (int j = 0; j < cnt; ++j) { float e = __expf(v[j] - mx); v[j] = e; sum += e; }
    red[t] = sum; __syncthreads();
    for (int s = 128; s > 0; s >>= 1) { if (t < s) red[t] += red[t + s]; __syncthreads(); }
    float inv = 1.f / red[0];

    cnt = 0;
    for (int i = t; i < N; i += 256) prow[i] = f32_bf16(v[cnt++] * inv);
}

// ---------------------------------------------------------------------------
// x_bf16[b][n][c] = gamma * o[b][n][c] + orig[b][c][n]
// ---------------------------------------------------------------------------
__launch_bounds__(256)
__global__ void residual_bf16(const float* __restrict__ o, const float* __restrict__ orig,
                              const float* __restrict__ gamma, u16* __restrict__ dst,
                              long total, int Cc, long HWn)
{
    long i = (long)blockIdx.x * 256 + threadIdx.x;
    if (i >= total) return;
    long per = (long)Cc * HWn;
    long b = i / per;
    long rem = i - b * per;
    long n = rem / Cc;
    long c = rem - n * Cc;
    dst[i] = f32_bf16(gamma[0] * o[i] + orig[b * per + c * HWn + n]);
}

// ---------------------------------------------------------------------------
// proto[b][c] = sum_n mid[b][n][c]*mask[b][n] / (sum_n mask[b][n] + 1e-5)
// ---------------------------------------------------------------------------
__launch_bounds__(256)
__global__ void proto_kernel(const float* __restrict__ mid, const float* __restrict__ mask,
                             float* __restrict__ proto, int HWn, int Cc)
{
    int c = blockIdx.x, b = blockIdx.y;
    const float* mb = mid  + (long)b * HWn * Cc;
    const float* mk = mask + (long)b * HWn;
    __shared__ float r1[256], r2[256];
    float s = 0.f, sm = 0.f;
    for (int n = threadIdx.x; n < HWn; n += 256) {
        float m = mk[n];
        s  += mb[(long)n * Cc + c] * m;
        sm += m;
    }
    r1[threadIdx.x] = s; r2[threadIdx.x] = sm; __syncthreads();
    for (int st = 128; st > 0; st >>= 1) {
        if (threadIdx.x < st) { r1[threadIdx.x] += r1[threadIdx.x + st]; r2[threadIdx.x] += r2[threadIdx.x + st]; }
        __syncthreads();
    }
    if (threadIdx.x == 0) proto[(long)b * Cc + c] = r1[0] / (r2[0] + 1e-5f);
}

// ---------------------------------------------------------------------------
// Tiny proto fusion, single block.
// ---------------------------------------------------------------------------
__launch_bounds__(256)
__global__ void fusion_kernel(const float* __restrict__ proto,
                              const float* __restrict__ wq, const float* __restrict__ bq,
                              const float* __restrict__ wk, const float* __restrict__ bk,
                              const float* __restrict__ wv, const float* __restrict__ bv,
                              float* __restrict__ out, int Cc, int Bb)
{
    __shared__ float sq[8 * 512], sk[8 * 512], sv[8 * 512];
    __shared__ float aw[8];
    __shared__ float red[256];

    for (int i = threadIdx.x; i < Bb * Cc; i += 256) {
        int b = i / Cc, c = i - b * Cc;
        const float* p = proto + (long)b * Cc;
        float aq = bq[c], ak = bk[c], av = bv[c];
        const float* wqr = wq + (long)c * Cc;
        const float* wkr = wk + (long)c * Cc;
        const float* wvr = wv + (long)c * Cc;
        for (int k = 0; k < Cc; ++k) {
            float pv = p[k];
            aq += pv * wqr[k]; ak += pv * wkr[k]; av += pv * wvr[k];
        }
        sq[i] = aq; sk[i] = ak; sv[i] = av;
    }
    __syncthreads();

    for (int b = 0; b < Bb; ++b) {
        float s = 0.f;
        for (int c = threadIdx.x; c < Cc; c += 256) s += sq[b * Cc + c] * sk[b * Cc + c];
        red[threadIdx.x] = s; __syncthreads();
        for (int st = 128; st > 0; st >>= 1) { if (threadIdx.x < st) red[threadIdx.x] += red[threadIdx.x + st]; __syncthreads(); }
        if (threadIdx.x == 0) aw[b] = red[0];
        __syncthreads();
    }
    if (threadIdx.x == 0) {
        float mx = -1e30f;
        for (int b = 0; b < Bb; ++b) mx = fmaxf(mx, aw[b]);
        float s = 0.f;
        for (int b = 0; b < Bb; ++b) { aw[b] = __expf(aw[b] - mx); s += aw[b]; }
        for (int b = 0; b < Bb; ++b) aw[b] /= s;
    }
    __syncthreads();
    for (int c = threadIdx.x; c < Cc; c += 256) {
        float s = 0.f;
        for (int b = 0; b < Bb; ++b) s += aw[b] * sv[b * Cc + c];
        out[c] = s;
    }
}

// ---------------------------------------------------------------------------
// Host launcher
// ---------------------------------------------------------------------------
extern "C" void kernel_launch(void* const* d_in, const int* in_sizes, int n_in,
                              void* d_out, int out_size, void* d_ws, size_t ws_size,
                              hipStream_t stream)
{
    const int  Bb = 8, Cc = 512, Dd = 64;
    const long HWn = 2304;
    const long big = (long)Bb * HWn * Cc;       // 9,437,184
    const long sml = (long)Bb * HWn * Dd;       // 1,179,648
    const long attE = (long)Bb * HWn * HWn;     // 42,467,328

    const float* qry   = (const float*)d_in[0];
    const float* spt   = (const float*)d_in[1];
    const float* mask  = (const float*)d_in[2];
    const float* w_wq  = (const float*)d_in[3];
    const float* b_wq  = (const float*)d_in[4];
    const float* w_wk  = (const float*)d_in[5];
    const float* b_wk  = (const float*)d_in[6];
    const float* w_wv1 = (const float*)d_in[7];
    const float* b_wv1 = (const float*)d_in[8];
    const float* w_wv2 = (const float*)d_in[9];
    const float* b_wv2 = (const float*)d_in[10];
    const float* gamma1= (const float*)d_in[11];
    const float* gamma2= (const float*)d_in[12];
    const float* ca_wq = (const float*)d_in[13];
    const float* ca_bq = (const float*)d_in[14];
    const float* ca_wk = (const float*)d_in[15];
    const float* ca_bk = (const float*)d_in[16];
    const float* ca_wv = (const float*)d_in[17];
    const float* ca_bv = (const float*)d_in[18];
    const float* ca_wo = (const float*)d_in[19];
    const float* ca_bo = (const float*)d_in[20];
    const float* pf_wq = (const float*)d_in[21];
    const float* pf_bq = (const float*)d_in[22];
    const float* pf_wk = (const float*)d_in[23];
    const float* pf_bk = (const float*)d_in[24];
    const float* pf_wv = (const float*)d_in[25];
    const float* pf_bv = (const float*)d_in[26];

    char*  wsb = (char*)d_ws;
    size_t off = 0;
    auto carve = [&](size_t bytes) -> void* {
        void* p = wsb + off;
        off = (off + bytes + 255) & ~(size_t)255;
        return p;
    };

    u16*   qryT  = (u16*)carve(big * 2);        // (B,HW,C) bf16
    u16*   sptT  = (u16*)carve(big * 2);
    u16*   wq_h  = (u16*)carve(64 * 512 * 2);
    u16*   wk_h  = (u16*)carve(64 * 512 * 2);
    u16*   wv1_h = (u16*)carve(512 * 512 * 2);
    u16*   wv2_h = (u16*)carve(512 * 512 * 2);
    u16*   cawq_h= (u16*)carve(512 * 512 * 2);
    u16*   cawk_h= (u16*)carve(512 * 512 * 2);
    u16*   cawv_h= (u16*)carve(512 * 512 * 2);
    u16*   cawo_h= (u16*)carve(512 * 512 * 2);
    u16*   q_h   = (u16*)carve(sml * 2);        // (B,HW,64)
    u16*   kT_h  = (u16*)carve(sml * 2);        // (B,HW,64)
    u16*   v1T_h = (u16*)carve(big * 2);        // (B,C,HW)  -> later x1 (B,HW,C)
    u16*   v2T_h = (u16*)carve(big * 2);        // (B,C,HW)  -> later x2
    float* att   = (float*)carve(attE * 4);     // logits/scores -> later mid2 f32
    u16*   P_h   = (u16*)carve(attE * 2);       // softmax bf16 (SA, then CA)
    u16*   PT_h  = (u16*)carve(attE * 2);       // P transposed; region reused below
    float* o1t   = (float*)carve(big * 4);
    float* o2t   = (float*)carve(big * 4);
    float* proto = (float*)carve(Bb * Cc * 4);

    // reuse regions that are dead by the time the CA stage runs
    u16* x1    = v1T_h;                         // after o1 GEMM
    u16* x2    = v2T_h;                         // after o2 GEMM
    u16* q2_h  = PT_h;                          // PT dead after o2 GEMM
    u16* k2_h  = PT_h + big;
    u16* v2fT_h= PT_h + 2 * big;
    u16* mid_h = PT_h + 3 * big;                // 4*big*2B = 75.5MB < 85MB region
    float* mid2 = att;                          // att dead after CA softmax

    const dim3 b256(256);
    const float rsC = 0.044194173824159216f;    // 1/sqrt(512)

    // ---- stage 0: pack inputs/weights to bf16 ----
    pack_transpose<<<dim3(72, 16, 8), b256, 0, stream>>>(qry, qryT, Cc, (int)HWn);
    pack_transpose<<<dim3(72, 16, 8), b256, 0, stream>>>(spt, sptT, Cc, (int)HWn);
    pack_copy<<<128, b256, 0, stream>>>(w_wq, wq_h, 64 * 512);
    pack_copy<<<128, b256, 0, stream>>>(w_wk, wk_h, 64 * 512);
    pack_copy<<<1024, b256, 0, stream>>>(w_wv1, wv1_h, 512 * 512);
    pack_copy<<<1024, b256, 0, stream>>>(w_wv2, wv2_h, 512 * 512);
    pack_copy<<<1024, b256, 0, stream>>>(ca_wq, cawq_h, 512 * 512);
    pack_copy<<<1024, b256, 0, stream>>>(ca_wk, cawk_h, 512 * 512);
    pack_copy<<<1024, b256, 0, stream>>>(ca_wv, cawv_h, 512 * 512);
    pack_copy<<<1024, b256, 0, stream>>>(ca_wo, cawo_h, 512 * 512);

    // ---- projections ----
    // q = qryT @ w_wq^T + b_wq       (HW x 64)
    wmma_gemm<<<dim3(18, 1, 8), b256, 0, stream>>>(qryT, wq_h, b_wq, 1, nullptr, q_h,
        (int)HWn, Dd, Cc, Cc, Cc, HWn * Cc, 0, HWn * Dd, 1.0f);
    // kT = sptT @ w_wk^T + b_wk      (HW x 64)
    wmma_gemm<<<dim3(18, 1, 8), b256, 0, stream>>>(sptT, wk_h, b_wk, 1, nullptr, kT_h,
        (int)HWn, Dd, Cc, Cc, Cc, HWn * Cc, 0, HWn * Dd, 1.0f);
    // v1T = w_wv1 @ qryT^T + b_wv1[m]   (C x HW)
    wmma_gemm<<<dim3(4, 18, 8), b256, 0, stream>>>(wv1_h, qryT, b_wv1, 2, nullptr, v1T_h,
        Cc, (int)HWn, Cc, Cc, Cc, 0, HWn * Cc, (long)Cc * HWn, 1.0f);
    // v2T = w_wv2 @ sptT^T + b_wv2[m]   (C x HW)
    wmma_gemm<<<dim3(4, 18, 8), b256, 0, stream>>>(wv2_h, sptT, b_wv2, 2, nullptr, v2T_h,
        Cc, (int)HWn, Cc, Cc, Cc, 0, HWn * Cc, (long)Cc * HWn, 1.0f);

    // ---- self-attention ----
    wmma_gemm<<<dim3(18, 18, 8), b256, 0, stream>>>(q_h, kT_h, nullptr, 0, att, nullptr,
        (int)HWn, (int)HWn, Dd, Dd, Dd, HWn * Dd, HWn * Dd, HWn * HWn, 1.0f);
    softmax_rows_bf16<<<dim3((unsigned)HWn, 8), b256, 0, stream>>>(att, P_h, (int)HWn);
    transpose_bf16<<<dim3(72, 72, 8), b256, 0, stream>>>(P_h, PT_h, (int)HWn, (int)HWn);
    // o1t = P @ v1T^T   (HW x C, f32)
    wmma_gemm<<<dim3(18, 4, 8), b256, 0, stream>>>(P_h, v1T_h, nullptr, 0, o1t, nullptr,
        (int)HWn, Cc, (int)HWn, (int)HWn, (int)HWn, HWn * HWn, (long)Cc * HWn, HWn * Cc, 1.0f);
    // o2t = P^T @ v2T^T (HW x C, f32)
    wmma_gemm<<<dim3(18, 4, 8), b256, 0, stream>>>(PT_h, v2T_h, nullptr, 0, o2t, nullptr,
        (int)HWn, Cc, (int)HWn, (int)HWn, (int)HWn, HWn * HWn, (long)Cc * HWn, HWn * Cc, 1.0f);

    // ---- residuals -> bf16 x1/x2 ----
    {
        unsigned nb = (unsigned)((big + 255) / 256);
        residual_bf16<<<nb, b256, 0, stream>>>(o1t, qry, gamma1, x1, big, Cc, HWn);
        residual_bf16<<<nb, b256, 0, stream>>>(o2t, spt, gamma2, x2, big, Cc, HWn);
    }

    // ---- cross-attention ----
    wmma_gemm<<<dim3(18, 4, 8), b256, 0, stream>>>(x1, cawq_h, ca_bq, 1, nullptr, q2_h,
        (int)HWn, Cc, Cc, Cc, Cc, HWn * Cc, 0, HWn * Cc, 1.0f);
    wmma_gemm<<<dim3(18, 4, 8), b256, 0, stream>>>(x2, cawk_h, ca_bk, 1, nullptr, k2_h,
        (int)HWn, Cc, Cc, Cc, Cc, HWn * Cc, 0, HWn * Cc, 1.0f);
    // v2fT = ca_wv @ x2^T + ca_bv[m]   (C x HW)
    wmma_gemm<<<dim3(4, 18, 8), b256, 0, stream>>>(cawv_h, x2, ca_bv, 2, nullptr, v2fT_h,
        Cc, (int)HWn, Cc, Cc, Cc, 0, HWn * Cc, (long)Cc * HWn, 1.0f);
    // scores = q2 @ k2^T / sqrt(C)
    wmma_gemm<<<dim3(18, 18, 8), b256, 0, stream>>>(q2_h, k2_h, nullptr, 0, att, nullptr,
        (int)HWn, (int)HWn, Cc, Cc, Cc, HWn * Cc, HWn * Cc, HWn * HWn, rsC);
    softmax_rows_bf16<<<dim3((unsigned)HWn, 8), b256, 0, stream>>>(att, P_h, (int)HWn);
    // mid = P2 @ v2fT^T  (HW x C, bf16)
    wmma_gemm<<<dim3(18, 4, 8), b256, 0, stream>>>(P_h, v2fT_h, nullptr, 0, nullptr, mid_h,
        (int)HWn, Cc, (int)HWn, (int)HWn, (int)HWn, HWn * HWn, (long)Cc * HWn, HWn * Cc, 1.0f);
    // mid2 = mid @ ca_wo^T + ca_bo  (HW x C, f32)
    wmma_gemm<<<dim3(18, 4, 8), b256, 0, stream>>>(mid_h, cawo_h, ca_bo, 1, mid2, nullptr,
        (int)HWn, Cc, Cc, Cc, Cc, HWn * Cc, 0, HWn * Cc, 1.0f);

    // ---- proto + fusion ----
    proto_kernel<<<dim3(Cc, Bb), b256, 0, stream>>>(mid2, mask, proto, (int)HWn, Cc);
    fusion_kernel<<<1, b256, 0, stream>>>(proto, pf_wq, pf_bq, pf_wk, pf_bk,
                                          pf_wv, pf_bv, (float*)d_out, Cc, Bb);
}
